// MambaBlock_29557964931128
// MI455X (gfx1250) — matrix operand order repaired and checked by reading, not compiled, not run on hardware
//
#include <hip/hip_runtime.h>
#include <hip/hip_bf16.h>

// ---------------- problem constants ----------------
#define BB        2
#define LL        2048
#define DMODEL    1024
#define DINNER    2048
#define DCONV     4
#define DTRANK    64
#define DSTATE    16
#define ROWS      (BB * LL)          // 4096 token rows
#define EPSV      1e-6f

typedef __bf16 bf16;
typedef __attribute__((ext_vector_type(16))) __bf16 v16bf;
typedef __attribute__((ext_vector_type(8)))  __bf16 v8bf;
typedef __attribute__((ext_vector_type(8)))  float  v8f;

__device__ __forceinline__ bf16 f2bf(float f) { return (bf16)f; }

__device__ __forceinline__ v16bf frag_cat(v8bf lo, v8bf hi) {
    v16bf r;
#pragma unroll
    for (int i = 0; i < 8; ++i) { r[i] = lo[i]; r[i + 8] = hi[i]; }
    return r;
}

// Epilogue modes (compile-time)
#define MODE_PLAIN    0   // C0[m,n] = acc
#define MODE_SPLIT    1   // n < Nsplit -> C0 fp32 ; else -> C1 bf16
#define MODE_SOFTPLUS 2   // C0[m,n] = softplus(acc + bias[n])
#define MODE_RESID    3   // C0[m,n] = acc + resid[m,n]

// ---------------- bf16 WMMA GEMM: C[M,N] = A[M,K] @ Wt[N,K]^T ----------------
// block = 256 thr = 8 waves; block tile = 128(M) x (NT*16)(N); wave tile 16 x NT*16.
// Grid guarantees every tile is in range; register double-buffered K pipeline.
template <int NT, int MODE>
__global__ void k_gemm_wmma(const bf16* __restrict__ A, const bf16* __restrict__ Wt,
                            int M, int N, int K,
                            float* __restrict__ C0, bf16* __restrict__ C1, int Nsplit,
                            const float* __restrict__ resid, const float* __restrict__ bias) {
    const int lane  = threadIdx.x & 31;
    const int wave  = threadIdx.x >> 5;
    const int mBase = blockIdx.x * 128 + wave * 16;
    const int nB0   = blockIdx.y * (NT * 16);
    const int mRow  = lane & 15;
    const int h     = lane >> 4;

    // Per-lane base pointers (A row-major [M,K]; Wt row-major [N,K])
    const bf16* aPtr = A + (size_t)(mBase + mRow) * K + h * 8;       // +k, +k+16
    const bf16* bPtr[NT];
#pragma unroll
    for (int t = 0; t < NT; ++t)
        bPtr[t] = Wt + (size_t)(nB0 + t * 16 + mRow) * K + h * 8;    // +k, +k+16

    v8f acc[NT];
#pragma unroll
    for (int t = 0; t < NT; ++t)
        acc[t] = (v8f){0.f, 0.f, 0.f, 0.f, 0.f, 0.f, 0.f, 0.f};

    // ---- software pipeline over K (32 per stage), register double-buffered ----
    v8bf alo, ahi, blo[NT], bhi[NT];
    alo = *(const v8bf*)(aPtr);
    ahi = *(const v8bf*)(aPtr + 16);
#pragma unroll
    for (int t = 0; t < NT; ++t) {
        blo[t] = *(const v8bf*)(bPtr[t]);
        bhi[t] = *(const v8bf*)(bPtr[t] + 16);
    }

    const int kIters = K >> 5;
    for (int i = 0; i < kIters - 1; ++i) {
        const int kn = (i + 1) << 5;
        // prefetch next chunk
        v8bf nalo = *(const v8bf*)(aPtr + kn);
        v8bf nahi = *(const v8bf*)(aPtr + kn + 16);
        v8bf nblo[NT], nbhi[NT];
#pragma unroll
        for (int t = 0; t < NT; ++t) {
            nblo[t] = *(const v8bf*)(bPtr[t] + kn);
            nbhi[t] = *(const v8bf*)(bPtr[t] + kn + 16);
        }
        // compute current chunk
        v16bf af = frag_cat(alo, ahi);
#pragma unroll
        for (int t = 0; t < NT; ++t) {
            v16bf bfm = frag_cat(blo[t], bhi[t]);
            acc[t] = __builtin_amdgcn_wmma_f32_16x16x32_bf16(
                false, af, false, bfm, (short)0, acc[t], false, false);
        }
        // rotate buffers
        alo = nalo; ahi = nahi;
#pragma unroll
        for (int t = 0; t < NT; ++t) { blo[t] = nblo[t]; bhi[t] = nbhi[t]; }
    }
    // final chunk
    {
        v16bf af = frag_cat(alo, ahi);
#pragma unroll
        for (int t = 0; t < NT; ++t) {
            v16bf bfm = frag_cat(blo[t], bhi[t]);
            acc[t] = __builtin_amdgcn_wmma_f32_16x16x32_bf16(
                false, af, false, bfm, (short)0, acc[t], false, false);
        }
    }

    // Epilogue: C element j of acc -> row m = mBase + 8*h + j, col n = nB0+t*16+(lane&15)
#pragma unroll
    for (int t = 0; t < NT; ++t) {
        int n = nB0 + t * 16 + mRow;
#pragma unroll
        for (int j = 0; j < 8; ++j) {
            int m = mBase + h * 8 + j;
            float v = acc[t][j];
            if constexpr (MODE == MODE_PLAIN) {
                C0[(size_t)m * N + n] = v;
            } else if constexpr (MODE == MODE_RESID) {
                C0[(size_t)m * N + n] = v + resid[(size_t)m * N + n];
            } else if constexpr (MODE == MODE_SPLIT) {
                if (n < Nsplit) C0[(size_t)m * Nsplit + n] = v;
                else            C1[(size_t)m * (N - Nsplit) + (n - Nsplit)] = f2bf(v);
            } else { // MODE_SOFTPLUS
                float z = v + bias[n];
                float sp = (z > 20.f) ? z : log1pf(__expf(z));
                C0[(size_t)m * N + n] = sp;
            }
        }
    }
}

// ---------------- weight transpose / bf16 convert ----------------
// W fp32 [K,N] -> Wt bf16 [N,K]
__global__ void k_transpose_bf16(const float* __restrict__ W, bf16* __restrict__ Wt,
                                 int K, int N) {
    int idx = blockIdx.x * blockDim.x + threadIdx.x;
    if (idx >= K * N) return;
    int k = idx / N, n = idx - k * N;
    Wt[(size_t)n * K + k] = f2bf(W[idx]);
}

// ---------------- RMSNorm -> u bf16 ----------------
__global__ void k_rmsnorm(const float* __restrict__ in, const float* __restrict__ scale,
                          bf16* __restrict__ u) {
    __shared__ float red[256];
    int row = blockIdx.x;
    const float* p = in + (size_t)row * DMODEL;
    float s = 0.f;
    for (int i = threadIdx.x; i < DMODEL; i += 256) { float v = p[i]; s += v * v; }
    red[threadIdx.x] = s;
    __syncthreads();
#pragma unroll
    for (int st = 128; st > 0; st >>= 1) {
        if (threadIdx.x < st) red[threadIdx.x] += red[threadIdx.x + st];
        __syncthreads();
    }
    float inv = rsqrtf(red[0] * (1.0f / DMODEL) + EPSV);
    for (int i = threadIdx.x; i < DMODEL; i += 256)
        u[(size_t)row * DMODEL + i] = f2bf(p[i] * inv * scale[i]);
}

// ---------------- depthwise conv along d (per (b,l) row) + SiLU, in place ----------------
// x[row, d] ; kernel ck[k*L + l] (shape (4,1,L)), SAME pad_left = 1
__global__ void k_conv_silu(float* __restrict__ x, const float* __restrict__ ck,
                            const float* __restrict__ cb) {
    __shared__ float s[DINNER];
    int row = blockIdx.x;
    int l   = row & (LL - 1);
    float* xr = x + (size_t)row * DINNER;
    for (int i = threadIdx.x; i < DINNER; i += 256) s[i] = xr[i];
    __syncthreads();
    float c0 = ck[l], c1 = ck[LL + l], c2 = ck[2 * LL + l], c3 = ck[3 * LL + l];
    float bb = cb[l];
    for (int d = threadIdx.x; d < DINNER; d += 256) {
        float acc = bb + s[d] * c1;
        if (d >= 1)          acc += s[d - 1] * c0;
        if (d + 1 < DINNER)  acc += s[d + 1] * c2;
        if (d + 2 < DINNER)  acc += s[d + 2] * c3;
        float sv = acc / (1.f + __expf(-acc));
        xr[d] = sv;
    }
}

// ---------------- extract delta_raw (x_dbl[:, :64]) as bf16 GEMM operand ----------------
__global__ void k_dr_bf16(const float* __restrict__ xdbl, bf16* __restrict__ dr) {
    int idx = blockIdx.x * blockDim.x + threadIdx.x;
    if (idx >= ROWS * DTRANK) return;
    int m = idx >> 6, r = idx & 63;
    dr[idx] = f2bf(xdbl[(size_t)m * (DTRANK + 2 * DSTATE) + r]);
}

// ---------------- selective scan + skip (x*D) + gate (silu(res)) -> yg bf16 ----------------
// one thread per (b,d); blocks of 256 consecutive d (never straddle b)
__global__ void k_scan(const float* __restrict__ delta, const float* __restrict__ x,
                       const float* __restrict__ xdbl, const bf16* __restrict__ resb,
                       const float* __restrict__ A_log, const float* __restrict__ Dv,
                       bf16* __restrict__ yg) {
    __shared__ float sBC[32];                    // 16 B then 16 C for current (b,l)
    int tid = blockIdx.x * 256 + threadIdx.x;    // 0 .. B*DINNER-1
    int b = tid / DINNER, d = tid - b * DINNER;
    float a[DSTATE], carry[DSTATE];
#pragma unroll
    for (int n = 0; n < DSTATE; ++n) {
        a[n] = -__expf(A_log[(size_t)d * DSTATE + n]);
        carry[n] = 0.f;
    }
    float Dd = Dv[d];
    for (int l = 0; l < LL; ++l) {
        size_t m = (size_t)b * LL + l;
        __syncthreads();
        if (threadIdx.x < 32)
            sBC[threadIdx.x] = xdbl[m * (DTRANK + 2 * DSTATE) + DTRANK + threadIdx.x];
        __syncthreads();
        float du  = delta[m * DINNER + d];
        float xv  = x[m * DINNER + d];
        float dux = du * xv;
        float y = 0.f;
#pragma unroll
        for (int n = 0; n < DSTATE; ++n) {
            carry[n] = __expf(du * a[n]) * carry[n] + dux * sBC[n];
            y += carry[n] * sBC[DSTATE + n];
        }
        float rv = (float)resb[m * DINNER + d];
        float g  = rv / (1.f + __expf(-rv));
        yg[m * DINNER + d] = f2bf((y + xv * Dd) * g);
    }
}

// ---------------- host side ----------------
static inline size_t alignUp(size_t v) { return (v + 255) & ~(size_t)255; }

extern "C" void kernel_launch(void* const* d_in, const int* in_sizes, int n_in,
                              void* d_out, int out_size, void* d_ws, size_t ws_size,
                              hipStream_t stream) {
    const float* inputs   = (const float*)d_in[0];   // (B,L,DMODEL)
    const float* nscale   = (const float*)d_in[1];   // (DMODEL)
    const float* W_in     = (const float*)d_in[2];   // (DMODEL, 2*DINNER)
    const float* convk    = (const float*)d_in[3];   // (4,1,L)
    const float* convb    = (const float*)d_in[4];   // (L)
    const float* W_x      = (const float*)d_in[5];   // (DMODEL, 96)
    const float* W_dt     = (const float*)d_in[6];   // (DTRANK, DINNER)
    const float* b_dt     = (const float*)d_in[7];   // (DINNER)
    const float* W_out    = (const float*)d_in[8];   // (DINNER, DMODEL)
    const float* A_log    = (const float*)d_in[9];   // (DINNER, DSTATE)
    const float* Dvec     = (const float*)d_in[10];  // (DINNER)
    float* out            = (float*)d_out;           // (B,L,DMODEL)

    const int NX = DTRANK + 2 * DSTATE;              // 96

    // workspace carve-up
    char* ws = (char*)d_ws;
    size_t o = 0;
    auto take = [&](size_t bytes) { size_t r = o; o += alignUp(bytes); return r; };
    bf16*  u      = (bf16*) (ws + take((size_t)ROWS * DMODEL * 2));        // 8 MB
    bf16*  WinT   = (bf16*) (ws + take((size_t)(2 * DINNER) * DMODEL * 2));// 8 MB
    bf16*  WxT    = (bf16*) (ws + take((size_t)NX * DMODEL * 2));
    bf16*  WdtT   = (bf16*) (ws + take((size_t)DINNER * DTRANK * 2));
    bf16*  WoutT  = (bf16*) (ws + take((size_t)DMODEL * DINNER * 2));      // 4 MB
    float* x      = (float*)(ws + take((size_t)ROWS * DINNER * 4));        // 32 MB
    bf16*  resb   = (bf16*) (ws + take((size_t)ROWS * DINNER * 2));        // 16 MB
    float* xdbl   = (float*)(ws + take((size_t)ROWS * NX * 4));
    bf16*  dr     = (bf16*) (ws + take((size_t)ROWS * DTRANK * 2));
    float* delta  = (float*)(ws + take((size_t)ROWS * DINNER * 4));        // 32 MB
    bf16*  yg     = (bf16*) (ws + take((size_t)ROWS * DINNER * 2));        // 16 MB
    (void)ws_size;

    // 1) weights -> bf16, (N,K) order
    {
        int n;
        n = DMODEL * 2 * DINNER;
        k_transpose_bf16<<<(n + 255) / 256, 256, 0, stream>>>(W_in,  WinT,  DMODEL, 2 * DINNER);
        n = DMODEL * NX;
        k_transpose_bf16<<<(n + 255) / 256, 256, 0, stream>>>(W_x,   WxT,   DMODEL, NX);
        n = DTRANK * DINNER;
        k_transpose_bf16<<<(n + 255) / 256, 256, 0, stream>>>(W_dt,  WdtT,  DTRANK, DINNER);
        n = DINNER * DMODEL;
        k_transpose_bf16<<<(n + 255) / 256, 256, 0, stream>>>(W_out, WoutT, DINNER, DMODEL);
    }

    // 2) RMSNorm -> u (bf16)
    k_rmsnorm<<<ROWS, 256, 0, stream>>>(inputs, nscale, u);

    // 3) xz = u @ W_in ; split x fp32 / res bf16   (N = 4096, NT=4)
    {
        dim3 g(ROWS / 128, (2 * DINNER) / 64);
        k_gemm_wmma<4, MODE_SPLIT><<<g, 256, 0, stream>>>(
            u, WinT, ROWS, 2 * DINNER, DMODEL, x, resb, DINNER, nullptr, nullptr);
    }

    // 4) x_dbl = u @ W_x (fp32)   (N = 96, NT=2, grid.y = 3)
    {
        dim3 g(ROWS / 128, NX / 32);
        k_gemm_wmma<2, MODE_PLAIN><<<g, 256, 0, stream>>>(
            u, WxT, ROWS, NX, DMODEL, xdbl, nullptr, 0, nullptr, nullptr);
    }

    // 5) depthwise conv along d + SiLU (in place on x)
    k_conv_silu<<<ROWS, 256, 0, stream>>>(x, convk, convb);

    // 6) delta = softplus(delta_raw @ W_dt + b_dt)   (N = 2048, NT=4)
    k_dr_bf16<<<(ROWS * DTRANK + 255) / 256, 256, 0, stream>>>(xdbl, dr);
    {
        dim3 g(ROWS / 128, DINNER / 64);
        k_gemm_wmma<4, MODE_SOFTPLUS><<<g, 256, 0, stream>>>(
            dr, WdtT, ROWS, DINNER, DTRANK, delta, nullptr, 0, nullptr, b_dt);
    }

    // 7) selective scan + skip + gate -> yg (bf16)
    k_scan<<<(BB * DINNER) / 256, 256, 0, stream>>>(delta, x, xdbl, resb,
                                                    A_log, Dvec, yg);

    // 8) out = yg @ W_out + inputs   (N = 1024, NT=4)
    {
        dim3 g(ROWS / 128, DMODEL / 64);
        k_gemm_wmma<4, MODE_RESID><<<g, 256, 0, stream>>>(
            yg, WoutT, ROWS, DMODEL, DINNER, out, nullptr, 0, inputs, nullptr);
    }
}
